// LSTMSOH_31696858645045
// MI455X (gfx1250) — matrix operand order repaired and checked by reading, not compiled, not run on hardware
//
#include <hip/hip_runtime.h>

#define HDIM 128
#define GDIM 512            // 4*H gates
#define BATCH 512
#define TSEQ 256
#define NPRED 50
#define MB 64               // batch rows per block
#define NTHREADS 512        // 16 waves (wave32)

typedef __attribute__((ext_vector_type(16))) _Float16 v16h;
typedef __attribute__((ext_vector_type(8)))  float    v8f;

union AFrag { v16h v; float4 q[2]; };

__device__ __forceinline__ float sigmoidf_(float x){ return 1.0f/(1.0f+__expf(-x)); }

// A fragment (16x32 f16), tile row-major with row stride HDIM.
// lanes 0-15: row=lane, K = kc*32 + {0..7, 16..23}
// lanes 16-31: row=lane-16, K = kc*32 + {8..15, 24..31}
__device__ __forceinline__ v16h frag_a(const _Float16* tile, int row0, int kc, int lane){
  int r  = lane & 15;
  int kh = (lane & 16) ? 8 : 0;
  const _Float16* p = tile + (row0 + r) * HDIM + kc * 32 + kh;
  AFrag f;
  f.q[0] = *(const float4*)(p);
  f.q[1] = *(const float4*)(p + 16);
  return f.v;
}

// B fragment (32x16 f16). W stored as (N=512, K=128) row-major, so B column n
// is W row n (contiguous). lanes 0-15: K = kc*32+{0..7,16..23}; lanes 16-31: +{8..15,24..31}.
__device__ __forceinline__ v16h frag_b(const _Float16* w, int ncol, int kc, int lane){
  int kh = (lane & 16) ? 8 : 0;
  const _Float16* p = w + ncol * HDIM + kc * 32 + kh;
  AFrag f;
  f.q[0] = *(const float4*)(p);
  f.q[1] = *(const float4*)(p + 16);
  return f.v;
}

// acc[s][g] += A(64x128 tile, this wave's mt) * W^T  for gate g, n-tile ntbase+s
// The opaque `lo` offset defeats LICM: without it the compiler hoists all 32
// weight fragments (256 VGPRs) out of the time loop and spills them to scratch,
// feeding every WMMA from scratch_load instead of ds_load.
__device__ __forceinline__ void gemm_step(const _Float16* atile, const _Float16* w,
                                          int mt, int ntbase, int lane, v8f acc[2][4]){
  int lo = 0;
  asm volatile("" : "+v"(lo));          // opaque 0: addresses unknown per call
  const _Float16* al = atile + lo;
  const _Float16* wl = w + lo;
  #pragma unroll
  for (int kc = 0; kc < HDIM/32; ++kc){
    v16h a = frag_a(al, mt*16, kc, lane);
    #pragma unroll
    for (int s = 0; s < 2; ++s){
      #pragma unroll
      for (int g = 0; g < 4; ++g){
        int ncol = g*HDIM + (ntbase + s)*16 + (lane & 15);
        v16h b = frag_b(wl, ncol, kc, lane);
        acc[s][g] = __builtin_amdgcn_wmma_f32_16x16x32_f16(
            false, a, false, b, (short)0, acc[s][g], false, false);
      }
    }
  }
}

__global__ void cvt_f32_to_f16(const float* __restrict__ src, _Float16* __restrict__ dst, int n){
  int i = (blockIdx.x * blockDim.x + threadIdx.x) * 4;
  if (i < n){
    float4 w = *(const float4*)(src + i);
    dst[i+0] = (_Float16)w.x; dst[i+1] = (_Float16)w.y;
    dst[i+2] = (_Float16)w.z; dst[i+3] = (_Float16)w.w;
  }
}

// One encoder LSTM layer; persistent time scan. Block owns MB batch rows.
template<bool LAYER0>
__global__ __launch_bounds__(NTHREADS, 1)
void lstm_enc_layer(const float* __restrict__ xg,          // layer0: x (B,T,4) f32
                    const _Float16* __restrict__ inseq,    // layers>0: (B,T,H) f16
                    _Float16* __restrict__ outseq,         // (B,T,H) f16
                    const float* __restrict__ wih,         // (512,din) f32
                    const float* __restrict__ whh,         // (512,128) f32
                    const float* __restrict__ bih,
                    const float* __restrict__ bhh,
                    float* __restrict__ h_state,           // (512,128) f32 final h
                    float* __restrict__ c_state)           // (512,128) f32 final c
{
  extern __shared__ char smem[];
  char* sp = smem;
  _Float16* Whh_l = (_Float16*)sp;              sp += (size_t)GDIM*HDIM*2;
  _Float16* Wih_l = nullptr;
  if (!LAYER0){ Wih_l = (_Float16*)sp;          sp += (size_t)GDIM*HDIM*2; }
  _Float16* htile = (_Float16*)sp;              sp += (size_t)MB*HDIM*2;
  _Float16* xtile = nullptr;
  float4*   xt32  = nullptr;
  float4*   wih0  = nullptr;
  if (LAYER0){ xt32 = (float4*)sp;              sp += (size_t)MB*16;
               wih0 = (float4*)sp;              sp += (size_t)GDIM*16; }
  else       { xtile = (_Float16*)sp;           sp += (size_t)MB*HDIM*2; }
  float* bias = (float*)sp;

  const int tid  = threadIdx.x;
  const int lane = tid & 31;
  const int wave = tid >> 5;
  const int mt     = wave & 3;        // 4 m-tiles of 16 rows
  const int ntbase = (wave >> 2) * 2; // 8 h n-tiles, 2 per wave
  const int b0 = blockIdx.x * MB;

  // stage weights (f32 -> f16) and bias into LDS
  for (int i = tid*4; i < GDIM*HDIM; i += NTHREADS*4){
    float4 w = *(const float4*)(whh + i);
    Whh_l[i+0]=(_Float16)w.x; Whh_l[i+1]=(_Float16)w.y;
    Whh_l[i+2]=(_Float16)w.z; Whh_l[i+3]=(_Float16)w.w;
    if (!LAYER0){
      float4 u = *(const float4*)(wih + i);
      Wih_l[i+0]=(_Float16)u.x; Wih_l[i+1]=(_Float16)u.y;
      Wih_l[i+2]=(_Float16)u.z; Wih_l[i+3]=(_Float16)u.w;
    }
  }
  if (LAYER0 && tid < GDIM) wih0[tid] = ((const float4*)wih)[tid]; // (512,4) rows
  if (tid < GDIM) bias[tid] = bih[tid] + bhh[tid];
  for (int i = tid; i < MB*HDIM; i += NTHREADS) htile[i] = (_Float16)0.f;

  v8f c[2];
  #pragma unroll
  for (int s = 0; s < 2; ++s)
    #pragma unroll
    for (int r = 0; r < 8; ++r) c[s][r] = 0.f;

  for (int t = 0; t < TSEQ; ++t){
    __syncthreads();                      // prev-step htile writes complete
    if (LAYER0){
      if (tid < MB){
        xt32[tid] = ((const float4*)xg)[(size_t)(b0 + tid)*TSEQ + t];
        if (t + 1 < TSEQ)
          __builtin_prefetch(&((const float4*)xg)[(size_t)(b0 + tid)*TSEQ + t + 1], 0, 3);
      }
    } else {
      int row = tid >> 3, ch = tid & 7;   // 512 thr * 32B = 16KB tile
      const _Float16* src = inseq + ((size_t)(b0+row)*TSEQ + t)*HDIM + ch*16;
      float4* dst = (float4*)(xtile + row*HDIM + ch*16);
      dst[0] = ((const float4*)src)[0];
      dst[1] = ((const float4*)src)[1];
      if (t + 1 < TSEQ) __builtin_prefetch(src + HDIM, 0, 3);
    }
    __syncthreads();

    v8f acc[2][4];
    #pragma unroll
    for (int s = 0; s < 2; ++s)
      #pragma unroll
      for (int g = 0; g < 4; ++g){
        float bv = bias[g*HDIM + (ntbase+s)*16 + (lane & 15)];
        #pragma unroll
        for (int r = 0; r < 8; ++r) acc[s][g][r] = bv;
      }

    if (LAYER0){
      // K=4 input contribution on VALU (WMMA would waste a K=32 pass)
      #pragma unroll
      for (int s = 0; s < 2; ++s)
        #pragma unroll
        for (int g = 0; g < 4; ++g){
          float4 wv = wih0[g*HDIM + (ntbase+s)*16 + (lane & 15)];
          #pragma unroll
          for (int r = 0; r < 8; ++r){
            int m = mt*16 + r + ((lane & 16) ? 8 : 0);
            float4 xv = xt32[m];
            acc[s][g][r] += xv.x*wv.x + xv.y*wv.y + xv.z*wv.z + xv.w*wv.w;
          }
        }
    } else {
      gemm_step(xtile, Wih_l, mt, ntbase, lane, acc);   // x_t @ wih^T
    }
    gemm_step(htile, Whh_l, mt, ntbase, lane, acc);     // h_{t-1} @ whh^T
    __syncthreads();                                     // all htile reads done

    // LSTM cell in registers; this wave owns all 4 gates of its (m,n) block
    #pragma unroll
    for (int s = 0; s < 2; ++s){
      int n = (ntbase+s)*16 + (lane & 15);
      #pragma unroll
      for (int r = 0; r < 8; ++r){
        int m = mt*16 + r + ((lane & 16) ? 8 : 0);
        float cc = sigmoidf_(acc[s][1][r]) * c[s][r]
                 + sigmoidf_(acc[s][0][r]) * tanhf(acc[s][2][r]);
        c[s][r] = cc;
        float hh = sigmoidf_(acc[s][3][r]) * tanhf(cc);
        htile[m*HDIM + n] = (_Float16)hh;
        outseq[((size_t)(b0+m)*TSEQ + t)*HDIM + n] = (_Float16)hh;
        if (t == TSEQ-1){
          h_state[(size_t)(b0+m)*HDIM + n] = hh;
          c_state[(size_t)(b0+m)*HDIM + n] = cc;
        }
      }
    }
  }
}

// Autoregressive decoder: 3 stacked cells + FC feedback, 50 steps.
// w16 slots: [0]=d0_whh [1]=d1_wih [2]=d1_whh [3]=d2_wih [4]=d2_whh (f16, (N,K))
__global__ __launch_bounds__(NTHREADS, 1)
void lstm_decoder(const float* __restrict__ xg,
                  const _Float16* __restrict__ w16,
                  const float* __restrict__ d0_wih,   // (512,4) f32
                  const float* __restrict__ b0i, const float* __restrict__ b0h,
                  const float* __restrict__ b1i, const float* __restrict__ b1h,
                  const float* __restrict__ b2i, const float* __restrict__ b2h,
                  const float* __restrict__ fcw, const float* __restrict__ fcb,
                  const float* __restrict__ h_state, const float* __restrict__ c_state,
                  float* __restrict__ out)
{
  extern __shared__ char smem[];
  char* sp = smem;
  _Float16* hl  = (_Float16*)sp;  sp += (size_t)3*MB*HDIM*2;   // h per layer
  float*   bias = (float*)sp;     sp += (size_t)3*GDIM*4;
  float4*  wih0 = (float4*)sp;    sp += (size_t)GDIM*16;
  float4*  xin  = (float4*)sp;    sp += (size_t)MB*16;
  float*   fcl  = (float*)sp;

  const int tid = threadIdx.x, lane = tid & 31, wave = tid >> 5;
  const int mt = wave & 3, ntbase = (wave >> 2) * 2;
  const int b0 = blockIdx.x * MB;

  if (tid < GDIM){
    bias[tid]          = b0i[tid] + b0h[tid];
    bias[GDIM + tid]   = b1i[tid] + b1h[tid];
    bias[2*GDIM + tid] = b2i[tid] + b2h[tid];
    wih0[tid] = ((const float4*)d0_wih)[tid];
  }
  if (tid < HDIM) fcl[tid] = fcw[tid];
  if (tid == 0)   fcl[HDIM] = fcb[0];
  if (tid < MB)   xin[tid] = ((const float4*)xg)[(size_t)(b0+tid)*TSEQ + (TSEQ-1)];
  for (int i = tid; i < 3*MB*HDIM; i += NTHREADS){
    int l = i / (MB*HDIM), j = i % (MB*HDIM);
    int m = j / HDIM, k = j % HDIM;
    hl[i] = (_Float16)h_state[(size_t)l*BATCH*HDIM + (size_t)(b0+m)*HDIM + k];
  }

  v8f c[3][2];
  #pragma unroll
  for (int l = 0; l < 3; ++l)
    #pragma unroll
    for (int s = 0; s < 2; ++s){
      int n = (ntbase+s)*16 + (lane & 15);
      #pragma unroll
      for (int r = 0; r < 8; ++r){
        int m = mt*16 + r + ((lane & 16) ? 8 : 0);
        c[l][s][r] = c_state[(size_t)l*BATCH*HDIM + (size_t)(b0+m)*HDIM + n];
      }
    }
  __syncthreads();

  for (int p = 0; p < NPRED; ++p){
    #pragma unroll
    for (int l = 0; l < 3; ++l){
      v8f acc[2][4];
      #pragma unroll
      for (int s = 0; s < 2; ++s)
        #pragma unroll
        for (int g = 0; g < 4; ++g){
          float bv = bias[l*GDIM + g*HDIM + (ntbase+s)*16 + (lane & 15)];
          #pragma unroll
          for (int r = 0; r < 8; ++r) acc[s][g][r] = bv;
        }
      if (l == 0){
        #pragma unroll
        for (int s = 0; s < 2; ++s)
          #pragma unroll
          for (int g = 0; g < 4; ++g){
            float4 wv = wih0[g*HDIM + (ntbase+s)*16 + (lane & 15)];
            #pragma unroll
            for (int r = 0; r < 8; ++r){
              int m = mt*16 + r + ((lane & 16) ? 8 : 0);
              float4 xv = xin[m];
              acc[s][g][r] += xv.x*wv.x + xv.y*wv.y + xv.z*wv.z + xv.w*wv.w;
            }
          }
        gemm_step(hl, w16, mt, ntbase, lane, acc);                              // h0 @ whh0^T
      } else {
        gemm_step(hl + (size_t)(l-1)*MB*HDIM, w16 + (size_t)(2*l-1)*GDIM*HDIM,
                  mt, ntbase, lane, acc);                                       // h_{l-1} @ wih^T
        gemm_step(hl + (size_t)l*MB*HDIM,     w16 + (size_t)(2*l)*GDIM*HDIM,
                  mt, ntbase, lane, acc);                                       // h_l @ whh^T
      }
      __syncthreads();
      #pragma unroll
      for (int s = 0; s < 2; ++s){
        int n = (ntbase+s)*16 + (lane & 15);
        #pragma unroll
        for (int r = 0; r < 8; ++r){
          int m = mt*16 + r + ((lane & 16) ? 8 : 0);
          float cc = sigmoidf_(acc[s][1][r]) * c[l][s][r]
                   + sigmoidf_(acc[s][0][r]) * tanhf(acc[s][2][r]);
          c[l][s][r] = cc;
          float hh = sigmoidf_(acc[s][3][r]) * tanhf(cc);
          hl[l*MB*HDIM + m*HDIM + n] = (_Float16)hh;
        }
      }
      __syncthreads();
    }
    // FC + autoregressive feedback
    if (tid < MB){
      const _Float16* hp = hl + 2*MB*HDIM + tid*HDIM;
      float s = fcl[HDIM];
      #pragma unroll 8
      for (int k = 0; k < HDIM; ++k) s += fcl[k] * (float)hp[k];
      out[(size_t)(b0+tid)*NPRED + p] = s;
      float4 xi; xi.x = s; xi.y = 0.f; xi.z = 0.f; xi.w = 0.f;
      xin[tid] = xi;
    }
    __syncthreads();
  }
}

extern "C" void kernel_launch(void* const* d_in, const int* in_sizes, int n_in,
                              void* d_out, int out_size, void* d_ws, size_t ws_size,
                              hipStream_t stream)
{
  (void)in_sizes; (void)n_in; (void)out_size; (void)ws_size;
  const float* x = (const float*)d_in[0];
  const float* e_wih[3] = {(const float*)d_in[1],  (const float*)d_in[5],  (const float*)d_in[9]};
  const float* e_whh[3] = {(const float*)d_in[2],  (const float*)d_in[6],  (const float*)d_in[10]};
  const float* e_bih[3] = {(const float*)d_in[3],  (const float*)d_in[7],  (const float*)d_in[11]};
  const float* e_bhh[3] = {(const float*)d_in[4],  (const float*)d_in[8],  (const float*)d_in[12]};
  const float* dw_ih[3] = {(const float*)d_in[13], (const float*)d_in[17], (const float*)d_in[21]};
  const float* dw_hh[3] = {(const float*)d_in[14], (const float*)d_in[18], (const float*)d_in[22]};
  const float* db_ih[3] = {(const float*)d_in[15], (const float*)d_in[19], (const float*)d_in[23]};
  const float* db_hh[3] = {(const float*)d_in[16], (const float*)d_in[20], (const float*)d_in[24]};
  const float* fcw = (const float*)d_in[25];
  const float* fcb = (const float*)d_in[26];

  char* ws = (char*)d_ws;
  _Float16* seqA = (_Float16*)ws;   ws += (size_t)BATCH*TSEQ*HDIM*2;   // 32 MB
  _Float16* seqB = (_Float16*)ws;   ws += (size_t)BATCH*TSEQ*HDIM*2;   // 32 MB
  _Float16* w16  = (_Float16*)ws;   ws += (size_t)5*GDIM*HDIM*2;       // 640 KB
  float* h_state = (float*)ws;      ws += (size_t)3*BATCH*HDIM*4;      // 768 KB
  float* c_state = (float*)ws;      ws += (size_t)3*BATCH*HDIM*4;      // 768 KB

  // decoder weights f32 -> f16 (5 matrices of 512x128)
  const float* srcs[5] = {dw_hh[0], dw_ih[1], dw_hh[1], dw_ih[2], dw_hh[2]};
  for (int i = 0; i < 5; ++i)
    cvt_f32_to_f16<<<GDIM*HDIM/1024, 256, 0, stream>>>(srcs[i], w16 + (size_t)i*GDIM*HDIM, GDIM*HDIM);

  size_t smem0 = (size_t)GDIM*HDIM*2   + MB*HDIM*2   + MB*16 + GDIM*16 + GDIM*4;   // ~155 KB
  size_t smem1 = (size_t)GDIM*HDIM*2*2 + MB*HDIM*2*2 + GDIM*4;                      // ~290 KB (CDNA5: 320KB/WGP)
  lstm_enc_layer<true ><<<BATCH/MB, NTHREADS, smem0, stream>>>(
      x, nullptr, seqA, e_wih[0], e_whh[0], e_bih[0], e_bhh[0], h_state, c_state);
  lstm_enc_layer<false><<<BATCH/MB, NTHREADS, smem1, stream>>>(
      nullptr, seqA, seqB, e_wih[1], e_whh[1], e_bih[1], e_bhh[1],
      h_state + BATCH*HDIM, c_state + BATCH*HDIM);
  lstm_enc_layer<false><<<BATCH/MB, NTHREADS, smem1, stream>>>(
      nullptr, seqB, seqA, e_wih[2], e_whh[2], e_bih[2], e_bhh[2],
      h_state + 2*BATCH*HDIM, c_state + 2*BATCH*HDIM);

  size_t smemd = (size_t)3*MB*HDIM*2 + 3*GDIM*4 + GDIM*16 + MB*16 + (HDIM+1)*4;     // ~64 KB
  lstm_decoder<<<BATCH/MB, NTHREADS, smemd, stream>>>(
      x, w16, dw_ih[0], db_ih[0], db_hh[0], db_ih[1], db_hh[1], db_ih[2], db_hh[2],
      fcw, fcb, h_state, c_state, (float*)d_out);
}
